// TransformerBlock_QKV_90984587199094
// MI455X (gfx1250) — matrix operand order repaired and checked by reading, not compiled, not run on hardware
//
#include <hip/hip_runtime.h>
#include <math.h>

// ---------------- problem constants ----------------
#define BATCH 8
#define DIM   64
#define HT    128
#define WD    128
#define HW    16384          // 128*128
#define HID   192            // 3*DIM
#define NHEAD 2

typedef __attribute__((ext_vector_type(16))) _Float16 v16h;
typedef __attribute__((ext_vector_type(8)))  float    v8f;

__device__ __forceinline__ float gelu_f(float x) {
    // exact erf GELU (torch nn.GELU default)
    return 0.5f * x * (1.0f + erff(x * 0.70710678118654752f));
}

// ---- WMMA fragment helpers (contiguous K order, shared by A and B) ----
// A 16x32 f16: lane<16 holds K {0..15}; lane>=16 holds K {16..31}
__device__ __forceinline__ v16h load_a_frag(const float* __restrict__ wr, int half) {
    float4 wa = *(const float4*)(wr + half * 16);
    float4 wb = *(const float4*)(wr + half * 16 + 4);
    float4 wc = *(const float4*)(wr + half * 16 + 8);
    float4 wd = *(const float4*)(wr + half * 16 + 12);
    v16h a;
    a[0]=(_Float16)wa.x; a[1]=(_Float16)wa.y; a[2]=(_Float16)wa.z; a[3]=(_Float16)wa.w;
    a[4]=(_Float16)wb.x; a[5]=(_Float16)wb.y; a[6]=(_Float16)wb.z; a[7]=(_Float16)wb.w;
    a[8]=(_Float16)wc.x; a[9]=(_Float16)wc.y; a[10]=(_Float16)wc.z; a[11]=(_Float16)wc.w;
    a[12]=(_Float16)wd.x; a[13]=(_Float16)wd.y; a[14]=(_Float16)wd.z; a[15]=(_Float16)wd.w;
    return a;
}
// B 32x16 f16: lane group selects K half (half*16 .. +15), col = lane&15; K strided
__device__ __forceinline__ v16h load_b_frag_strided(const float* __restrict__ xc) {
    v16h bf;
    #pragma unroll
    for (int j = 0; j < 16; ++j) bf[j] = (_Float16)xc[(size_t)j * HW];
    return bf;
}
// B 32x16 f16 with contiguous K (for Gram: K = spatial index)
__device__ __forceinline__ v16h load_b_frag_contig(const float* __restrict__ kc) {
    float4 k0v = *(const float4*)(kc);
    float4 k1v = *(const float4*)(kc + 4);
    float4 k2v = *(const float4*)(kc + 8);
    float4 k3v = *(const float4*)(kc + 12);
    v16h bf;
    bf[0]=(_Float16)k0v.x; bf[1]=(_Float16)k0v.y; bf[2]=(_Float16)k0v.z; bf[3]=(_Float16)k0v.w;
    bf[4]=(_Float16)k1v.x; bf[5]=(_Float16)k1v.y; bf[6]=(_Float16)k1v.z; bf[7]=(_Float16)k1v.w;
    bf[8]=(_Float16)k2v.x; bf[9]=(_Float16)k2v.y; bf[10]=(_Float16)k2v.z; bf[11]=(_Float16)k2v.w;
    bf[12]=(_Float16)k3v.x; bf[13]=(_Float16)k3v.y; bf[14]=(_Float16)k3v.z; bf[15]=(_Float16)k3v.w;
    return bf;
}

// =====================================================================
// LayerNorm over channel dim (C=64) of NCHW, per pixel.
// =====================================================================
__global__ void ln_kernel(const float* __restrict__ x,
                          const float* __restrict__ w,
                          const float* __restrict__ b,
                          float* __restrict__ out) {
    size_t g = (size_t)blockIdx.x * blockDim.x + threadIdx.x;   // < BATCH*HW
    int bb = (int)(g / HW);
    int p  = (int)(g % HW);
    const float* base = x + (size_t)bb * DIM * HW + p;
    float s = 0.f, s2 = 0.f;
    for (int c = 0; c < DIM; ++c) {
        float v = base[(size_t)c * HW];
        s += v; s2 += v * v;
    }
    float mu  = s * (1.0f / DIM);
    float var = s2 * (1.0f / DIM) - mu * mu;
    float inv = rsqrtf(var + 1e-5f);
    float* ob = out + (size_t)bb * DIM * HW + p;
    for (int c = 0; c < DIM; ++c) {
        float v = base[(size_t)c * HW];
        ob[(size_t)c * HW] = (v - mu) * inv * w[c] + b[c];
    }
}

// =====================================================================
// Pointwise (1x1) conv as WMMA GEMM: out[b,o,p] = sum_c W[o,c]*in[b,c,p] + bias[o]
// One wave = 32(out-ch) x 32(pixel) macro-tile = 2x2 WMMA tiles; each A/B
// fragment feeds two v_wmma's.  CIN and residual-ACC are compile-time.
// =====================================================================
template<int CIN, bool ACC>
__global__ void pw_wmma_kernel(const float* __restrict__ in,
                               const float* __restrict__ w,
                               const float* __restrict__ bias,
                               float* __restrict__ out,
                               int Cout) {
    int lane = threadIdx.x;          // 0..31
    int lr   = lane & 15;
    int half = lane >> 4;
    int P0 = blockIdx.x * 32;
    int O0 = blockIdx.y * 32;
    int b  = blockIdx.z;
    const float* xb = in + (size_t)b * CIN * HW;

    v8f acc00 = {}, acc01 = {}, acc10 = {}, acc11 = {};
    #pragma unroll 2
    for (int k0 = 0; k0 < CIN; k0 += 32) {
        const float* wr0 = w + (size_t)(O0 + lr) * CIN + k0;
        const float* wr1 = w + (size_t)(O0 + 16 + lr) * CIN + k0;
        v16h a0 = load_a_frag(wr0, half);
        v16h a1 = load_a_frag(wr1, half);
        const float* xc = xb + (size_t)(k0 + half * 16) * HW + P0;
        v16h b0 = load_b_frag_strided(xc + lr);
        v16h b1 = load_b_frag_strided(xc + 16 + lr);
        acc00 = __builtin_amdgcn_wmma_f32_16x16x32_f16(false, a0, false, b0, (short)0, acc00, false, false);
        acc01 = __builtin_amdgcn_wmma_f32_16x16x32_f16(false, a0, false, b1, (short)0, acc01, false, false);
        acc10 = __builtin_amdgcn_wmma_f32_16x16x32_f16(false, a1, false, b0, (short)0, acc10, false, false);
        acc11 = __builtin_amdgcn_wmma_f32_16x16x32_f16(false, a1, false, b1, (short)0, acc11, false, false);
    }
    // ---- D store: VGPR r -> (M=r+8*half, N=lane&15) per 16x16 tile ----
    #pragma unroll
    for (int mt = 0; mt < 2; ++mt) {
        v8f* accs = mt ? &acc10 : &acc00;   // [mt][0], [mt][1] adjacent below
        v8f t0 = mt ? acc10 : acc00;
        v8f t1 = mt ? acc11 : acc01;
        (void)accs;
        #pragma unroll
        for (int r = 0; r < 8; ++r) {
            int o = O0 + mt * 16 + r + 8 * half;
            float bs = bias[o];
            size_t row = ((size_t)b * Cout + o) * HW;
            size_t i0 = row + (P0 + lr);
            size_t i1 = row + (P0 + 16 + lr);
            float v0 = t0[r] + bs;
            float v1 = t1[r] + bs;
            if (ACC) { v0 += out[i0]; v1 += out[i1]; }
            out[i0] = v0;
            out[i1] = v1;
        }
    }
}

// =====================================================================
// Depthwise 3x3, stride 1, SAME; optional GELU.
// =====================================================================
__global__ void dw3_kernel(const float* __restrict__ in,
                           const float* __restrict__ wdw,
                           const float* __restrict__ bias,
                           float* __restrict__ out,
                           int C, int do_gelu) {
    int p = blockIdx.x * blockDim.x + threadIdx.x;  // 0..HW-1
    int c = blockIdx.y;
    int b = blockIdx.z;
    int h = p >> 7, w = p & 127;
    const float* src = in + ((size_t)b * C + c) * HW;
    float acc = bias[c];
    #pragma unroll
    for (int dy = -1; dy <= 1; ++dy)
        #pragma unroll
        for (int dx = -1; dx <= 1; ++dx) {
            int hh = h + dy, ww = w + dx;
            if (hh >= 0 && hh < HT && ww >= 0 && ww < WD)
                acc += wdw[c * 9 + (dy + 1) * 3 + (dx + 1)] * src[hh * WD + ww];
        }
    if (do_gelu) acc = gelu_f(acc);
    out[((size_t)b * C + c) * HW + p] = acc;
}

// =====================================================================
// Fused concat(A[64ch view], B[64ch]) -> dw3 -> GELU : 128 output channels.
// =====================================================================
__global__ void dwcat_kernel(const float* __restrict__ inA, int aC, int aOff,
                             const float* __restrict__ inB,
                             const float* __restrict__ wdw,
                             const float* __restrict__ bias,
                             float* __restrict__ out, int do_gelu) {
    int p = blockIdx.x * blockDim.x + threadIdx.x;
    int c = blockIdx.y;                 // 0..127
    int b = blockIdx.z;
    int h = p >> 7, w = p & 127;
    const float* src = (c < DIM)
        ? inA + ((size_t)b * aC + aOff + c) * HW
        : inB + ((size_t)b * DIM + (c - DIM)) * HW;
    float acc = bias[c];
    #pragma unroll
    for (int dy = -1; dy <= 1; ++dy)
        #pragma unroll
        for (int dx = -1; dx <= 1; ++dx) {
            int hh = h + dy, ww = w + dx;
            if (hh >= 0 && hh < HT && ww >= 0 && ww < WD)
                acc += wdw[c * 9 + (dy + 1) * 3 + (dx + 1)] * src[hh * WD + ww];
        }
    if (do_gelu) acc = gelu_f(acc);
    out[((size_t)b * 2 * DIM + c) * HW + p] = acc;
}

// =====================================================================
// FFN gate: h[384ch] -> g[o] = gelu(dw(h,o)+b[o]) * (dw(h,o+192)+b[o+192])
// =====================================================================
__global__ void ffn_gate_kernel(const float* __restrict__ hbuf,
                                const float* __restrict__ wdw,
                                const float* __restrict__ bias,
                                float* __restrict__ g) {
    int p = blockIdx.x * blockDim.x + threadIdx.x;
    int o = blockIdx.y;                 // 0..191
    int b = blockIdx.z;
    int h = p >> 7, w = p & 127;
    const float* s1 = hbuf + ((size_t)b * 2 * HID + o) * HW;
    const float* s2 = hbuf + ((size_t)b * 2 * HID + o + HID) * HW;
    float r1 = bias[o], r2 = bias[o + HID];
    #pragma unroll
    for (int dy = -1; dy <= 1; ++dy)
        #pragma unroll
        for (int dx = -1; dx <= 1; ++dx) {
            int hh = h + dy, ww = w + dx;
            if (hh >= 0 && hh < HT && ww >= 0 && ww < WD) {
                int tap = (dy + 1) * 3 + (dx + 1);
                r1 += wdw[o * 9 + tap] * s1[hh * WD + ww];
                r2 += wdw[(o + HID) * 9 + tap] * s2[hh * WD + ww];
            }
        }
    g[((size_t)b * HID + o) * HW + p] = gelu_f(r1) * r2;
}

// =====================================================================
// Row L2 norm over N=HW for 64 logical channels (view with aC/aOff).
// =====================================================================
__global__ void rownorm_kernel(const float* __restrict__ t, int C, int off,
                               float* __restrict__ norms) {
    __shared__ float red[256];
    int r = blockIdx.x;                 // b*64 + c
    int b = r >> 6, c = r & 63;
    const float* base = t + ((size_t)b * C + off + c) * HW;
    float s = 0.f;
    for (int i = threadIdx.x; i < HW; i += 256) {
        float v = base[i];
        s += v * v;
    }
    red[threadIdx.x] = s;
    __syncthreads();
    for (int st = 128; st > 0; st >>= 1) {
        if (threadIdx.x < st) red[threadIdx.x] += red[threadIdx.x + st];
        __syncthreads();
    }
    if (threadIdx.x == 0) norms[r] = sqrtf(red[0]);
}

// =====================================================================
// Gram matrix S[c,d] = sum_n q[c,n]*k[d,n] per (batch, head) via WMMA.
// One wave per 16x16 tile of the 32x32 output; K loop over HW (contiguous).
// =====================================================================
__global__ void gram_wmma_kernel(const float* __restrict__ q, int qC, int qOff,
                                 const float* __restrict__ k, int kC, int kOff,
                                 float* __restrict__ S) {
    int lane = threadIdx.x;
    int lr = lane & 15, half = lane >> 4;
    int bh = blockIdx.z;                 // b*2 + head
    int b  = bh >> 1, hd = bh & 1;
    int M0 = blockIdx.x * 16;
    int D0 = blockIdx.y * 16;
    const float* qr = q + ((size_t)b * qC + qOff + hd * 32 + M0 + lr) * HW;
    const float* kr = k + ((size_t)b * kC + kOff + hd * 32 + D0 + lr) * HW;

    v8f acc8 = {};
    #pragma unroll 2
    for (int n0 = 0; n0 < HW; n0 += 32) {
        v16h a  = load_a_frag(qr + n0, half);
        v16h bf = load_b_frag_contig(kr + n0 + half * 16);
        acc8 = __builtin_amdgcn_wmma_f32_16x16x32_f16(
            false, a, false, bf, (short)0, acc8, false, false);
    }
    #pragma unroll
    for (int r = 0; r < 8; ++r) {
        int m = M0 + r + 8 * half;
        int d = D0 + lr;
        S[((size_t)bh * 32 + m) * 32 + d] = acc8[r];
    }
}

// =====================================================================
// Scale by 1/(|q||k|)*temp, top-k threshold (>= thr kept), masked softmax.
// =====================================================================
__global__ void topk_softmax_kernel(const float* __restrict__ S,
                                    const float* __restrict__ nq,
                                    const float* __restrict__ nk,
                                    const float* __restrict__ temp,
                                    const int* __restrict__ topk,
                                    float* __restrict__ Wout) {
    int bh = blockIdx.x;
    int b = bh >> 1, hd = bh & 1;
    int c = threadIdx.x;                // 0..31
    const float* Sr = S + ((size_t)bh * 32 + c) * 32;
    float tq = fmaxf(nq[b * 64 + hd * 32 + c], 1e-12f);
    float t  = temp[hd];
    float a[32];
    #pragma unroll
    for (int d = 0; d < 32; ++d) {
        float tk = fmaxf(nk[b * 64 + hd * 32 + d], 1e-12f);
        a[d] = Sr[d] / (tq * tk) * t;
    }
    int K = *topk;
    if (K > 32) K = 32;
    if (K < 1)  K = 1;
    // thr = K-th largest = min over { a[d] : #{e: a[e] > a[d]} < K }
    float thr = INFINITY, rowmax = -INFINITY;
    #pragma unroll
    for (int d = 0; d < 32; ++d) {
        int cnt = 0;
        #pragma unroll
        for (int e = 0; e < 32; ++e) cnt += (a[e] > a[d]) ? 1 : 0;
        if (cnt < K) thr = fminf(thr, a[d]);
        rowmax = fmaxf(rowmax, a[d]);
    }
    float Z = 0.f;
    #pragma unroll
    for (int d = 0; d < 32; ++d)
        if (a[d] >= thr) Z += expf(a[d] - rowmax);
    float invZ = 1.0f / Z;
    #pragma unroll
    for (int d = 0; d < 32; ++d)
        Wout[((size_t)bh * 32 + c) * 32 + d] =
            (a[d] >= thr) ? expf(a[d] - rowmax) * invZ : 0.f;
}

// =====================================================================
// out[c,n] = sum_d attn[c,d] * v[d,n]   (attn 32x32 cached in LDS)
// =====================================================================
__global__ void attnv_kernel(const float* __restrict__ Wm,
                             const float* __restrict__ v, int vC, int vOff,
                             float* __restrict__ out) {
    __shared__ float ws[32][32];
    int bh = blockIdx.z;
    int b = bh >> 1, hd = bh & 1;
    const float* Wp = Wm + (size_t)bh * 1024;
    for (int i = threadIdx.x; i < 1024; i += blockDim.x)
        ws[i >> 5][i & 31] = Wp[i];
    __syncthreads();
    int p = blockIdx.x * blockDim.x + threadIdx.x;
    float vv[32];
    #pragma unroll
    for (int d = 0; d < 32; ++d)
        vv[d] = v[((size_t)b * vC + vOff + hd * 32 + d) * HW + p];
    #pragma unroll 4
    for (int c = 0; c < 32; ++c) {
        float acc = 0.f;
        #pragma unroll
        for (int d = 0; d < 32; ++d) acc += ws[c][d] * vv[d];
        out[((size_t)b * DIM + hd * 32 + c) * HW + p] = acc;
    }
}

// =====================================================================
// Host-side orchestration
// =====================================================================
extern "C" void kernel_launch(void* const* d_in, const int* in_sizes, int n_in,
                              void* d_out, int out_size, void* d_ws, size_t ws_size,
                              hipStream_t stream) {
    (void)in_sizes; (void)n_in; (void)ws_size;
    // ---- inputs ----
    const float* x        = (const float*)d_in[0];
    const float* feature1 = (const float*)d_in[1];
    const float* feature2 = (const float*)d_in[2];
    const float* feature3 = (const float*)d_in[3];
    const int*   topk     = (const int*)  d_in[4];
    // ---- params (setup_inputs dict order) ----
    const float* norm1_w   = (const float*)d_in[5];
    const float* norm1_b   = (const float*)d_in[6];
    const float* norm2_w   = (const float*)d_in[7];
    const float* norm2_b   = (const float*)d_in[8];
    const float* q_temp    = (const float*)d_in[9];
    const float* q_qkv_w   = (const float*)d_in[10];
    const float* q_qkv_b   = (const float*)d_in[11];
    const float* q_qkvdw_w = (const float*)d_in[12];
    const float* q_qkvdw_b = (const float*)d_in[13];
    const float* q_qdw_w   = (const float*)d_in[14];
    const float* q_qdw_b   = (const float*)d_in[15];
    const float* q_qpw_w   = (const float*)d_in[16];
    const float* q_qpw_b   = (const float*)d_in[17];
    const float* q_proj_w  = (const float*)d_in[18];
    const float* q_proj_b  = (const float*)d_in[19];
    const float* kv_temp   = (const float*)d_in[20];
    const float* kv_qkv_w  = (const float*)d_in[21];
    const float* kv_qkv_b  = (const float*)d_in[22];
    const float* kv_qkvdw_w= (const float*)d_in[23];
    const float* kv_qkvdw_b= (const float*)d_in[24];
    const float* kv_kdw_w  = (const float*)d_in[25];
    const float* kv_kdw_b  = (const float*)d_in[26];
    const float* kv_kpw_w  = (const float*)d_in[27];
    const float* kv_kpw_b  = (const float*)d_in[28];
    const float* kv_vdw_w  = (const float*)d_in[29];
    const float* kv_vdw_b  = (const float*)d_in[30];
    const float* kv_vpw_w  = (const float*)d_in[31];
    const float* kv_vpw_b  = (const float*)d_in[32];
    const float* kv_proj_w = (const float*)d_in[33];
    const float* kv_proj_b = (const float*)d_in[34];
    const float* ffn_in_w  = (const float*)d_in[35];
    const float* ffn_in_b  = (const float*)d_in[36];
    const float* ffn_dw_w  = (const float*)d_in[37];
    const float* ffn_dw_b  = (const float*)d_in[38];
    const float* ffn_out_w = (const float*)d_in[39];
    const float* ffn_out_b = (const float*)d_in[40];
    const float* fuse_dw_w = (const float*)d_in[41];
    const float* fuse_dw_b = (const float*)d_in[42];
    const float* fuse_pw_w = (const float*)d_in[43];
    const float* fuse_pw_b = (const float*)d_in[44];

    float* out = (float*)d_out;

    // ---- workspace arena (U = one [8,64,128,128] fp32 tensor) ----
    const size_t U = (size_t)BATCH * DIM * HW;   // 8388608 floats
    float* ws   = (float*)d_ws;
    float* xn   = ws;             // 1U : LN output (reused for LN2)
    float* t3a  = ws + 1 * U;     // 3U : pw 192ch out  (also FFN big6 = 6U span)
    float* t3b  = ws + 4 * U;     // 3U : dw 192ch out
    float* t2   = t3a;            // alias: 2U concat-dw-gelu out
    float* q2s  = ws + 3 * U;     // 1U : q2 / k2 (t3a's third unit)
    float* t2b  = ws + 7 * U;     // 2U : second concat-dw-gelu out
    float* att  = ws + 9 * U;     // 1U : attn output / v2
    float* aq   = ws + 10 * U;    // 1U
    float* akv  = ws + 11 * U;    // 1U
    float* gbuf = ws + 12 * U;    // 3U : FFN gate output
    float* sm   = ws + 15 * U;    // small scalars
    float* nq1 = sm;          float* nk1 = sm + 512;
    float* nq2 = sm + 1024;   float* nk2 = sm + 1536;
    float* S1  = sm + 2048;   float* S2  = S1 + 16384;
    float* W1  = S2 + 16384;  float* W2  = W1 + 16384;

    dim3 blk32(32), blk256(256);

    // ---- Phase 1: LN1 ----
    ln_kernel<<<dim3(BATCH * HW / 256), blk256, 0, stream>>>(x, norm1_w, norm1_b, xn);

    // ---- Phase 2: Attention_Query branch ----
    pw_wmma_kernel<64,  false><<<dim3(HW/32, 6, BATCH), blk32, 0, stream>>>(xn, q_qkv_w, q_qkv_b, t3a, 3*DIM);
    dw3_kernel<<<dim3(HW/256, 192, BATCH), blk256, 0, stream>>>(t3a, q_qkvdw_w, q_qkvdw_b, t3b, 192, 0);
    dwcat_kernel<<<dim3(HW/256, 128, BATCH), blk256, 0, stream>>>(t3b, 192, 0, feature1, q_qdw_w, q_qdw_b, t2, 1);
    pw_wmma_kernel<128, false><<<dim3(HW/32, 2, BATCH), blk32, 0, stream>>>(t2, q_qpw_w, q_qpw_b, q2s, DIM);
    rownorm_kernel<<<dim3(BATCH * 64), blk256, 0, stream>>>(q2s, 64, 0, nq1);
    rownorm_kernel<<<dim3(BATCH * 64), blk256, 0, stream>>>(t3b, 192, 64, nk1);
    gram_wmma_kernel<<<dim3(2, 2, BATCH * NHEAD), blk32, 0, stream>>>(q2s, 64, 0, t3b, 192, 64, S1);
    topk_softmax_kernel<<<dim3(BATCH * NHEAD), blk32, 0, stream>>>(S1, nq1, nk1, q_temp, topk, W1);
    attnv_kernel<<<dim3(HW/256, 1, BATCH * NHEAD), blk256, 0, stream>>>(W1, t3b, 192, 128, att);
    pw_wmma_kernel<64,  false><<<dim3(HW/32, 2, BATCH), blk32, 0, stream>>>(att, q_proj_w, q_proj_b, aq, DIM);

    // ---- Phase 3: Attention_Key_Value branch ----
    pw_wmma_kernel<64,  false><<<dim3(HW/32, 6, BATCH), blk32, 0, stream>>>(xn, kv_qkv_w, kv_qkv_b, t3a, 3*DIM);
    dw3_kernel<<<dim3(HW/256, 192, BATCH), blk256, 0, stream>>>(t3a, kv_qkvdw_w, kv_qkvdw_b, t3b, 192, 0);
    dwcat_kernel<<<dim3(HW/256, 128, BATCH), blk256, 0, stream>>>(t3b, 192, 64, feature2, kv_kdw_w, kv_kdw_b, t2, 1);
    pw_wmma_kernel<128, false><<<dim3(HW/32, 2, BATCH), blk32, 0, stream>>>(t2, kv_kpw_w, kv_kpw_b, q2s /*k2*/, DIM);
    dwcat_kernel<<<dim3(HW/256, 128, BATCH), blk256, 0, stream>>>(t3b, 192, 128, feature3, kv_vdw_w, kv_vdw_b, t2b, 1);
    pw_wmma_kernel<128, false><<<dim3(HW/32, 2, BATCH), blk32, 0, stream>>>(t2b, kv_vpw_w, kv_vpw_b, att /*v2*/, DIM);
    rownorm_kernel<<<dim3(BATCH * 64), blk256, 0, stream>>>(t3b, 192, 0, nq2);
    rownorm_kernel<<<dim3(BATCH * 64), blk256, 0, stream>>>(q2s, 64, 0, nk2);
    gram_wmma_kernel<<<dim3(2, 2, BATCH * NHEAD), blk32, 0, stream>>>(t3b, 192, 0, q2s, 64, 0, S2);
    topk_softmax_kernel<<<dim3(BATCH * NHEAD), blk32, 0, stream>>>(S2, nq2, nk2, kv_temp, topk, W2);
    attnv_kernel<<<dim3(HW/256, 1, BATCH * NHEAD), blk256, 0, stream>>>(W2, att, 64, 0, t2 /*att2 (1U)*/);
    pw_wmma_kernel<64,  false><<<dim3(HW/32, 2, BATCH), blk32, 0, stream>>>(t2, kv_proj_w, kv_proj_b, akv, DIM);

    // ---- Phase 4: fuse + residual (d_out = x + fused) ----
    dwcat_kernel<<<dim3(HW/256, 128, BATCH), blk256, 0, stream>>>(aq, 64, 0, akv, fuse_dw_w, fuse_dw_b, t2b, 1);
    hipMemcpyAsync(out, x, (size_t)out_size * sizeof(float), hipMemcpyDeviceToDevice, stream);
    pw_wmma_kernel<128, true><<<dim3(HW/32, 2, BATCH), blk32, 0, stream>>>(t2b, fuse_pw_w, fuse_pw_b, out, DIM);

    // ---- Phase 5: gated FFN + residual ----
    ln_kernel<<<dim3(BATCH * HW / 256), blk256, 0, stream>>>(out, norm2_w, norm2_b, xn);
    pw_wmma_kernel<64,  false><<<dim3(HW/32, 12, BATCH), blk32, 0, stream>>>(xn, ffn_in_w, ffn_in_b, t3a /*6U span*/, 2*HID);
    ffn_gate_kernel<<<dim3(HW/256, HID, BATCH), blk256, 0, stream>>>(t3a, ffn_dw_w, ffn_dw_b, gbuf);
    pw_wmma_kernel<192, true><<<dim3(HW/32, 2, BATCH), blk32, 0, stream>>>(gbuf, ffn_out_w, ffn_out_b, out, DIM);
}
